// BBBHetRegModel_26929444946732
// MI455X (gfx1250) — matrix operand order repaired and checked by reading, not compiled, not run on hardware
//
#include <hip/hip_runtime.h>
#include <hip/hip_bf16.h>
#include <math.h>

typedef __attribute__((ext_vector_type(16))) _Float16 v16h;
typedef __attribute__((ext_vector_type(8)))  float    v8f;

#define N_TOTAL 131072
#define HID     1024
#define PRIOR1  4.0f
#define PRIOR2  0.0703125f   /* 2.25 / sqrt(1024) */
#define MIN_STD 1e-5f

__device__ __forceinline__ float softplusf(float v) {
    // numerically stable softplus: max(v,0) + log1p(exp(-|v|))
    return fmaxf(v, 0.0f) + log1pf(expf(-fabsf(v)));
}

__device__ __forceinline__ float kl_term(float mu, float sigma, float prior) {
    float r = sigma / prior;
    float q = mu / prior;
    return 0.5f * (2.0f * logf(prior / sigma) - 1.0f + r * r + q * q);
}

// ---------------------------------------------------------------------------
// Main kernel: one wave per 16-row tile; 32 x v_wmma_f32_16x16x32_f16 over K.
// ---------------------------------------------------------------------------
__global__ __launch_bounds__(256) void bbb_main_kernel(
    const float* __restrict__ x,
    const float* __restrict__ W1_mu, const float* __restrict__ W1_rho,
    const float* __restrict__ b1_mu, const float* __restrict__ b1_rho,
    const float* __restrict__ W2_mu, const float* __restrict__ W2_rho,
    const float* __restrict__ b2_mu, const float* __restrict__ b2_rho,
    const float* __restrict__ eps_W1, const float* __restrict__ eps_b1,
    const float* __restrict__ eps_W2, const float* __restrict__ eps_b2,
    float* __restrict__ out)
{
    __shared__ float    sW1[HID];         // reparameterized layer-1 weights (f32)
    __shared__ float    sB1[HID];         // reparameterized layer-1 bias (f32)
    __shared__ _Float16 sW2p[3 * HID];    // rows 0,1 = reparam W2 (f16); row 2 = zeros
    __shared__ float    sB2[2];

    const int t = threadIdx.x;

    // Stage reparameterized weights into LDS (~14KB of the 320KB WGP pool).
    for (int j = t; j < HID; j += 256) {
        sW1[j] = fmaf(softplusf(W1_rho[j]), eps_W1[j], W1_mu[j]);
        sB1[j] = fmaf(softplusf(b1_rho[j]), eps_b1[j], b1_mu[j]);
    }
    for (int j = t; j < 2 * HID; j += 256) {
        sW2p[j] = (_Float16)fmaf(softplusf(W2_rho[j]), eps_W2[j], W2_mu[j]);
    }
    for (int j = t; j < HID; j += 256) {
        sW2p[2 * HID + j] = (_Float16)0.0f;   // zero row for padding columns 2..15
    }
    if (t < 2) sB2[t] = fmaf(softplusf(b2_rho[t]), eps_b2[t], b2_mu[t]);
    __syncthreads();

    const int wave = t >> 5;          // wave32
    const int lane = t & 31;
    const int m    = lane & 15;       // column index within 16x16 tile
    const int g    = lane >> 4;       // lane half-group
    const int tile = blockIdx.x * 8 + wave;
    const int n0   = tile * 16;

    // Each lane m holds x for matrix row m of this tile (lanes 16-31 mirror).
    const float xm = x[n0 + m];

    // Hoisted per-lane LDS bases.
    const int brow = (m < 2) ? m : 2;                   // zero row for cols >= 2
    const _Float16* __restrict__ bsrc = &sW2p[brow * HID + g * 8];
    const float* __restrict__ w1a = &sW1[g * 8];        // halves 0..7 :  k = kk + g*8 + h
    const float* __restrict__ w1b = &sW1[16 + g * 8];   // halves 8..15: k = kk + 16 + g*8 + h
    const float* __restrict__ b1a = &sB1[g * 8];
    const float* __restrict__ b1b = &sB1[16 + g * 8];

    v8f acc = {};
#pragma unroll 2
    for (int kk = 0; kk < HID; kk += 32) {
        // ---- A fragment: A[m][k] = relu(x[m]*W1[k] + b1[k]), f32 -> f16.
        // 16-bit A 16x32 layout: half h -> k = (h/8)*16 + g*8 + (h%8)
        v16h a;
#pragma unroll
        for (int h = 0; h < 8; ++h) {
            float v0 = fmaxf(fmaf(xm, w1a[kk + h], b1a[kk + h]), 0.0f);
            float v1 = fmaxf(fmaf(xm, w1b[kk + h], b1b[kk + h]), 0.0f);
            a[h]     = (_Float16)v0;
            a[h + 8] = (_Float16)v1;
        }

        // ---- B fragment (unconditional; cols >= 2 read the zero row).
        // 16-bit B 32x16 layout: half h -> k = (h/8)*16 + g*8 + (h%8), col = lane%16
        v16h b;
#pragma unroll
        for (int h = 0; h < 16; ++h) b[h] = bsrc[kk + (h & 7) + (h >> 3) * 16];

        // D = A x B + D  (f32 accumulate). EXEC is all-ones here.
        acc = __builtin_amdgcn_wmma_f32_16x16x32_f16(
            false, a, false, b, (short)0, acc, false, false);
    }

    // ---- Writeback. D layout: VGPR r holds D[g*8 + r][m]. Cols 0/1 useful.
    if (m < 2) {
        const float bias    = sB2[m];
        const int   rowBase = n0 + g * 8;
        if (m == 0) {
#pragma unroll
            for (int r = 0; r < 8; ++r)
                out[rowBase + r] = acc[r] + bias;
        } else {
#pragma unroll
            for (int r = 0; r < 8; ++r)
                out[N_TOTAL + rowBase + r] = MIN_STD + softplusf(acc[r] + bias);
        }
    }
}

// ---------------------------------------------------------------------------
// KL kernel: single block, tree reduction over ~4K parameter terms.
// ---------------------------------------------------------------------------
__global__ __launch_bounds__(1024) void bbb_kl_kernel(
    const float* __restrict__ W1_mu, const float* __restrict__ W1_rho,
    const float* __restrict__ b1_mu, const float* __restrict__ b1_rho,
    const float* __restrict__ W2_mu, const float* __restrict__ W2_rho,
    const float* __restrict__ b2_mu, const float* __restrict__ b2_rho,
    float* __restrict__ out)
{
    __shared__ float red[1024];
    const int t = threadIdx.x;

    float kl = 0.0f;
    // Layer 1: W1 (1024x1) and b1 (1024), prior sigma1
    kl += kl_term(W1_mu[t], softplusf(W1_rho[t]), PRIOR1);
    kl += kl_term(b1_mu[t], softplusf(b1_rho[t]), PRIOR1);
    // Layer 2: W2 (2x1024), prior sigma2
    for (int j = t; j < 2 * HID; j += 1024)
        kl += kl_term(W2_mu[j], softplusf(W2_rho[j]), PRIOR2);
    // Layer 2 bias (2), prior sigma2
    if (t < 2)
        kl += kl_term(b2_mu[t], softplusf(b2_rho[t]), PRIOR2);

    red[t] = kl;
    __syncthreads();
    for (int s = 512; s > 0; s >>= 1) {
        if (t < s) red[t] += red[t + s];
        __syncthreads();
    }
    if (t == 0) out[2 * N_TOTAL] = red[0];
}

extern "C" void kernel_launch(void* const* d_in, const int* in_sizes, int n_in,
                              void* d_out, int out_size, void* d_ws, size_t ws_size,
                              hipStream_t stream) {
    (void)in_sizes; (void)n_in; (void)d_ws; (void)ws_size; (void)out_size;
    const float* x      = (const float*)d_in[0];
    const float* W1_mu  = (const float*)d_in[1];
    const float* W1_rho = (const float*)d_in[2];
    const float* b1_mu  = (const float*)d_in[3];
    const float* b1_rho = (const float*)d_in[4];
    const float* W2_mu  = (const float*)d_in[5];
    const float* W2_rho = (const float*)d_in[6];
    const float* b2_mu  = (const float*)d_in[7];
    const float* b2_rho = (const float*)d_in[8];
    const float* eps_W1 = (const float*)d_in[9];
    const float* eps_b1 = (const float*)d_in[10];
    const float* eps_W2 = (const float*)d_in[11];
    const float* eps_b2 = (const float*)d_in[12];
    float* out = (float*)d_out;

    // 8192 row-tiles of 16, 8 waves (tiles) per 256-thread block.
    bbb_main_kernel<<<N_TOTAL / 16 / 8, 256, 0, stream>>>(
        x, W1_mu, W1_rho, b1_mu, b1_rho, W2_mu, W2_rho, b2_mu, b2_rho,
        eps_W1, eps_b1, eps_W2, eps_b2, out);

    bbb_kl_kernel<<<1, 1024, 0, stream>>>(
        W1_mu, W1_rho, b1_mu, b1_rho, W2_mu, W2_rho, b2_mu, b2_rho, out);
}
